// LaplaceConv1d_24421184045146
// MI455X (gfx1250) — matrix operand order repaired and checked by reading, not compiled, not run on hardware
//
#include <hip/hip_runtime.h>
#include <hip/hip_bf16.h>
#include <math.h>

#define MODES  32
#define IN_CH  32
#define OUT_CH 32
#define SIZE   2048
#define BATCH  8

typedef __attribute__((ext_vector_type(16))) __bf16 v16bf;
typedef __attribute__((ext_vector_type(8)))  float  v8f;

// ---- workspace byte offsets (total ~1.38 MB) ----
#define OFF_XMEAN 0                                   // 32 f32
#define OFF_A     128                                 // 32 f32 : a_k = poles/(S-1)
#define OFF_PF    256                                 // 32 f32 : pole filter
#define OFF_R     384                                 // 1 int  : causal half-width (samples)
#define OFF_SCALE 512                                 // 32*2048 f32 : pole_filter/row_norm
#define OFF_XT    (512 + MODES*SIZE*4)                // 8*2048*32 bf16 : x^T [b][s][c]
#define OFF_WFRAG (OFF_XT + BATCH*SIZE*IN_CH*2)       // 32*2*32*16 bf16 : A-fragment weights

// native f32 -> bf16 conversion (RNE in hardware on gfx1250; avoids the
// integer round-and-shift VALU chains the software version emitted)
__device__ __forceinline__ __bf16 f2bf(float f) { return (__bf16)f; }

// ---------------- K0: channel means of x over (batch, grid) ----------------
__global__ void k_xmean(const float* __restrict__ x, char* __restrict__ wsb) {
  const int c = blockIdx.x;
  __shared__ float red[256];
  float s = 0.f;
  for (int b = 0; b < BATCH; ++b) {
    const float* row = x + (b * IN_CH + c) * SIZE;
    for (int i = threadIdx.x; i < SIZE; i += 256) s += row[i];
  }
  red[threadIdx.x] = s;
  __syncthreads();
  for (int off = 128; off > 0; off >>= 1) {
    if (threadIdx.x < off) red[threadIdx.x] += red[threadIdx.x + off];
    __syncthreads();
  }
  if (threadIdx.x == 0)
    ((float*)(wsb + OFF_XMEAN))[c] = red[0] / (float)(BATCH * SIZE);
}

// ---------------- K1: poles, causal radius, closed-form row norms, weight frags ----
__global__ void k_prep(const float* __restrict__ c_star, const float* __restrict__ dt_star,
                       const float* __restrict__ wla, const float* __restrict__ wph,
                       const float* __restrict__ log_poles, const float* __restrict__ pole_w,
                       const float* __restrict__ pole_b, char* __restrict__ wsb) {
  float* xmean = (float*)(wsb + OFF_XMEAN);
  float* A     = (float*)(wsb + OFF_A);
  float* PF    = (float*)(wsb + OFF_PF);
  int*   Rp    = (int*)  (wsb + OFF_R);
  float* SC    = (float*)(wsb + OFF_SCALE);
  __bf16* WF   = (__bf16*)(wsb + OFF_WFRAG);
  const int tid = threadIdx.x;

  if (tid < MODES) {
    float dot = pole_b[tid];
    for (int c = 0; c < IN_CH; ++c) dot += xmean[c] * pole_w[c * MODES + tid];
    float v  = log_poles[tid] + 0.1f * tanhf(dot);
    float sp = (v > 20.f) ? v : log1pf(__expf(v));          // softplus
    sp = fminf(fmaxf(sp, 0.1f), 100.f);                     // pole clip
    A[tid] = sp / (float)(SIZE - 1);                        // per-sample decay rate
    float kn = (float)tid / (float)(MODES - 1);
    PF[tid] = __expf(-4.0f * kn * kn);                      // FILTER_STRENGTH=4
  }
  if (tid == 0) {
    float mx = c_star[0];
    for (int i = 1; i < BATCH; ++i) mx = fmaxf(mx, c_star[i]);
    *Rp = (int)floorf(mx * dt_star[0] * (float)(SIZE - 1)); // causal band half-width
  }
  __syncthreads();
  const int R = *Rp;

  // row_norm via geometric series: 1 + g(min(t,R)) + g(min(S-1-t,R)), g(D)=Σ_{d=1..D} e^{-a d}
  for (int idx = tid; idx < MODES * SIZE; idx += 256) {
    int k = idx >> 11, t = idx & (SIZE - 1);
    float a = A[k];
    int Dm = min(t, R), Dp = min(SIZE - 1 - t, R);
    float denom = -expm1f(-a);
    float ea = __expf(-a);
    float gm = (ea - __expf(-a * (float)(Dm + 1))) / denom;
    float gp = (ea - __expf(-a * (float)(Dp + 1))) / denom;
    SC[idx] = PF[k] / fmaxf(1.f + gm + gp, 1e-8f);
  }

  // weights[k,c,o] = sigmoid(wla)*cos(wph), scattered into WMMA A-fragment lane order:
  // frag[k][ot][lane][j] : o = ot*16+(lane&15), c = (lane>>4)*16 + j
  for (int idx = tid; idx < MODES * IN_CH * OUT_CH; idx += 256) {
    int k = idx >> 10, r = idx & 1023, ot = r >> 9, r2 = r & 511;
    int lane = r2 >> 4, j = r2 & 15;
    int o = ot * 16 + (lane & 15);
    int c = ((lane >> 4) << 4) + j;
    int wi = (k * IN_CH + c) * OUT_CH + o;
    float amp = 1.f / (1.f + __expf(-wla[wi]));
    WF[idx] = f2bf(amp * __cosf(wph[wi]));
  }
}

// ---------------- K2: x -> bf16, transposed to [b][s][c] for B-fragment loads ------
__global__ void k_xT(const float* __restrict__ x, char* __restrict__ wsb) {
  __bf16* xT = (__bf16*)(wsb + OFF_XT);
  int idx = blockIdx.x * 256 + threadIdx.x;      // b*65536 + s*32 + c
  int b = idx >> 16, rem = idx & 65535;
  int s = rem >> 5, c = rem & 31;
  xT[idx] = f2bf(x[(b * IN_CH + c) * SIZE + s]);
}

// ---------------- K3: fused dual-WMMA main kernel --------------------------------
// block = (b, 64-wide t tile); 8 waves: wave = (o-half)*4 + t-subtile
// stage1 (waves 0..3): z[o, s0..s0+31] = W_k^T x  -> bf16 in LDS
// stage2 (all):        out^T[t,o] += Exp(k,t,s)·z^T   (band-generated A fragment)
__global__ void __launch_bounds__(256) k_main(const char* __restrict__ wsb,
                                              float* __restrict__ out) {
  __shared__ __align__(32) __bf16 zt[OUT_CH * 32];   // [o][s_loc]

  const float* SC = (const float*)(wsb + OFF_SCALE);
  const float* A  = (const float*)(wsb + OFF_A);
  const char*  xT = wsb + OFF_XT;
  const char*  WF = wsb + OFF_WFRAG;
  const int R = *(const int*)(wsb + OFF_R);

  const int b     = blockIdx.x >> 5;
  const int t0blk = (blockIdx.x & 31) * 64;
  const int wave  = threadIdx.x >> 5;
  const int lane  = threadIdx.x & 31;
  const int t0w   = t0blk + (wave & 3) * 16;
  const int o0    = (wave >> 2) * 16;
  const int t_row = t0w + (lane & 15);
  const int hi    = lane >> 4;

  const int stlo = max(0, t0blk - R) >> 5;
  const int sthi = min(SIZE - 1, t0blk + 63 + R) >> 5;

  const int ot = (wave >> 1) & 1;   // stage-1 roles (only waves 0..3 execute)
  const int st = wave & 1;

  v8f acc = {};
  for (int k = 0; k < MODES; ++k) {
    const float a_k   = A[k];
    const float scale = SC[k * SIZE + t_row];      // pole_filter/row_norm, per output row
    const v16bf a1 = *(const v16bf*)(WF + ((((k * 2 + ot) * 32) + lane) << 5));

    for (int sti = stlo; sti <= sthi; ++sti) {
      const int s0 = sti << 5;
      __syncthreads();
      if (wave < 4) {
        const int scol = s0 + st * 16 + (lane & 15);
        const char* bp = xT + (((b * SIZE + scol) << 6) + (hi << 5));
        const v16bf b1 = *(const v16bf*)bp;
        __builtin_prefetch(bp + (32 << 6), 0, 1);  // next s-tile row -> global_prefetch
        v8f z = {};
        z = __builtin_amdgcn_wmma_f32_16x16x32_bf16(false, a1, false, b1,
                                                    (short)0, z, false, false);
        const int base = (ot * 16) * 32 + st * 16 + (lane & 15);
        #pragma unroll
        for (int r = 0; r < 8; ++r) {
          int m = hi ? (r + 8) : r;
          zt[base + m * 32] = f2bf(z[r]);          // ds_store_b16
        }
      }
      __syncthreads();

      // B2: z^T fragment (K=s contiguous 32B run in LDS row)
      const v16bf b2 = *(const v16bf*)(&zt[(o0 + (lane & 15)) * 32 + (hi << 4)]);

      // A2: band-limited Laplace kernel generated in registers (v_exp_f32)
      v16bf a2;
      #pragma unroll
      for (int j = 0; j < 16; ++j) {
        int s_loc = (hi << 4) + j;
        int d = t_row - (s0 + s_loc); d = d < 0 ? -d : d;
        float v = (d <= R) ? scale * __expf(-a_k * (float)d) : 0.0f;
        a2[j] = f2bf(v);
      }
      acc = __builtin_amdgcn_wmma_f32_16x16x32_bf16(false, a2, false, b2,
                                                    (short)0, acc, false, false);
    }
  }

  #pragma unroll
  for (int r = 0; r < 8; ++r) {
    int m = hi ? (r + 8) : r;
    int t = t0w + m;
    int o = o0 + (lane & 15);
    out[(b * OUT_CH + o) * SIZE + t] = acc[r];
  }
}

extern "C" void kernel_launch(void* const* d_in, const int* in_sizes, int n_in,
                              void* d_out, int out_size, void* d_ws, size_t ws_size,
                              hipStream_t stream) {
  (void)in_sizes; (void)n_in; (void)out_size; (void)ws_size;
  const float* x         = (const float*)d_in[0];
  const float* c_star    = (const float*)d_in[1];
  const float* dt_star   = (const float*)d_in[2];
  /* d_in[3] = dx_star (unused by forward) */
  const float* wla       = (const float*)d_in[4];
  const float* wph       = (const float*)d_in[5];
  const float* log_poles = (const float*)d_in[6];
  const float* pole_w    = (const float*)d_in[7];
  const float* pole_b    = (const float*)d_in[8];
  char*  wsb = (char*)d_ws;
  float* out = (float*)d_out;

  k_xmean<<<IN_CH, 256, 0, stream>>>(x, wsb);
  k_prep <<<1,     256, 0, stream>>>(c_star, dt_star, wla, wph, log_poles, pole_w, pole_b, wsb);
  k_xT   <<<(BATCH * SIZE * IN_CH) / 256, 256, 0, stream>>>(x, wsb);
  k_main <<<BATCH * 32, 256, 0, stream>>>(wsb, out);
}
